// GMNN_16638703305267
// MI455X (gfx1250) — matrix operand order repaired and checked by reading, not compiled, not run on hardware
//
#include <hip/hip_runtime.h>
#include <hip/hip_fp16.h>
#include <math.h>

#define A_ATOMS   30000
#define NPAIRS    1200000
#define NSPEC     119
#define NRAD      5
#define NBAS      7
#define KP1       384          // padded feature count (360 -> 384) for GEMM1
#define NFEAT     360
#define NU        512          // hidden units
#define MOMSZ     100          // unique moment floats per atom: 5*(1+3+6+10)

typedef __attribute__((ext_vector_type(16))) _Float16 v16h;
typedef __attribute__((ext_vector_type(8)))  float    v8f;

// symmetry index tables: full 3x3 -> 6 unique, full 3x3x3 -> 10 unique
__constant__ int c_sym2[9]  = {0,1,2, 1,3,4, 2,4,5};
__constant__ int c_sym3[27] = {0,1,2,1,3,4,2,4,5,
                               1,3,4,3,6,7,4,7,8,
                               2,4,5,4,7,8,5,8,9};

__global__ void zero_f32(float* __restrict__ p, int n) {
    int t = blockIdx.x * blockDim.x + threadIdx.x;
    if (t < n) p[t] = 0.0f;
}

// ---------------- Phase 1: pair scatter into unique moments --------------
__global__ void pair_kernel(const float* __restrict__ R, const int* __restrict__ Z,
                            const int* __restrict__ idx, const float* __restrict__ W,
                            float* __restrict__ mom) {
    int p = blockIdx.x * blockDim.x + threadIdx.x;
    if (p >= NPAIRS) return;
    int ai = idx[p];
    int aj = idx[NPAIRS + p];

    float dx = R[aj*3+0] - R[ai*3+0];
    float dy = R[aj*3+1] - R[ai*3+1];
    float dz = R[aj*3+2] - R[ai*3+2];
    float dr = sqrtf(dx*dx + dy*dy + dz*dz + 1e-12f);
    float inv = 1.0f / (dr + 1e-5f);
    float nx = dx*inv, ny = dy*inv, nz = dz*inv;

    // gaussian radial basis
    const float betta    = 49.0f / 36.0f;           // NBAS^2 / RMAX^2
    const float rad_norm = 0.96481436f;             // (2*betta/pi)^0.25
    float basis[NBAS];
#pragma unroll
    for (int b = 0; b < NBAS; b++) {
        float sh = 0.5f + (5.5f / 7.0f) * (float)b;
        float d  = dr - sh;
        basis[b] = rad_norm * __expf(-betta * d * d);
    }
    float cut = (dr < 6.0f) ? 0.5f * (__cosf(3.14159265358979f * dr * (1.0f/6.0f)) + 1.0f) : 0.0f;
    cut *= 0.37796447300923f;                       // 1/sqrt(NBAS)

    int zi = Z[ai], zj = Z[aj];
    const float* wrow = W + (size_t)(zi * NSPEC + zj) * (NRAD * NBAS);
    float rad[NRAD];
#pragma unroll
    for (int k = 0; k < NRAD; k++) {
        float s = 0.0f;
#pragma unroll
        for (int b = 0; b < NBAS; b++) s += wrow[k*NBAS + b] * basis[b];
        rad[k] = s * cut;
    }

    float xx = nx*nx, xy = nx*ny, xz = nx*nz, yy = ny*ny, yz = ny*nz, zz = nz*nz;
    float* base = mom + (size_t)ai * MOMSZ;
#pragma unroll
    for (int k = 0; k < NRAD; k++) {
        float rk = rad[k];
        atomicAdd(base + k, rk);                              // m0
        atomicAdd(base + 5 + k*3 + 0, rk*nx);                 // m1
        atomicAdd(base + 5 + k*3 + 1, rk*ny);
        atomicAdd(base + 5 + k*3 + 2, rk*nz);
        atomicAdd(base + 20 + k*6 + 0, rk*xx);                // m2 unique
        atomicAdd(base + 20 + k*6 + 1, rk*xy);
        atomicAdd(base + 20 + k*6 + 2, rk*xz);
        atomicAdd(base + 20 + k*6 + 3, rk*yy);
        atomicAdd(base + 20 + k*6 + 4, rk*yz);
        atomicAdd(base + 20 + k*6 + 5, rk*zz);
        atomicAdd(base + 50 + k*10 + 0, rk*xx*nx);            // m3 unique
        atomicAdd(base + 50 + k*10 + 1, rk*xx*ny);
        atomicAdd(base + 50 + k*10 + 2, rk*xx*nz);
        atomicAdd(base + 50 + k*10 + 3, rk*xy*ny);
        atomicAdd(base + 50 + k*10 + 4, rk*xy*nz);
        atomicAdd(base + 50 + k*10 + 5, rk*xz*nz);
        atomicAdd(base + 50 + k*10 + 6, rk*yy*ny);
        atomicAdd(base + 50 + k*10 + 7, rk*yy*nz);
        atomicAdd(base + 50 + k*10 + 8, rk*yz*nz);
        atomicAdd(base + 50 + k*10 + 9, rk*zz*nz);
    }
}

// ---------------- Phase 2: invariant contractions -> 360 features -------
#define CBLK 64
__global__ void contract_kernel(const float* __restrict__ mom, _Float16* __restrict__ gm) {
    __shared__ float sm[CBLK * 195];                 // 195 = 15(m1)+45(m2)+135(m3)
    int a = blockIdx.x * CBLK + threadIdx.x;
    if (a >= A_ATOMS) return;
    float* my = &sm[threadIdx.x * 195];              // 195 % 64 == 3 -> conflict-free
    const float* g = mom + (size_t)a * MOMSZ;

    float m0[5];
#pragma unroll
    for (int k = 0; k < 5; k++) m0[k] = g[k];
    for (int k = 0; k < 5; k++) {
#pragma unroll
        for (int c = 0; c < 3; c++) my[k*3 + c] = g[5 + k*3 + c];
#pragma unroll
        for (int e = 0; e < 9; e++)  my[15 + k*9  + e] = g[20 + k*6  + c_sym2[e]];
#pragma unroll
        for (int e = 0; e < 27; e++) my[60 + k*27 + e] = g[50 + k*10 + c_sym3[e]];
    }

    _Float16* out = gm + (size_t)a * KP1;
    int f = 0;
    // m0
    for (int k = 0; k < 5; k++) out[f++] = (_Float16)m0[k];
    // c1 = m1.m1
    for (int r = 0; r < 5; r++) for (int s = r; s < 5; s++) {
        float acc = 0.0f;
#pragma unroll
        for (int c = 0; c < 3; c++) acc += my[r*3+c] * my[s*3+c];
        out[f++] = (_Float16)acc;
    }
    // c2 = m2:m2
    for (int r = 0; r < 5; r++) for (int s = r; s < 5; s++) {
        float acc = 0.0f;
#pragma unroll
        for (int e = 0; e < 9; e++) acc += my[15+r*9+e] * my[15+s*9+e];
        out[f++] = (_Float16)acc;
    }
    // c3 = m3:m3
    for (int r = 0; r < 5; r++) for (int s = r; s < 5; s++) {
        float acc = 0.0f;
#pragma unroll
        for (int e = 0; e < 27; e++) acc += my[60+r*27+e] * my[60+s*27+e];
        out[f++] = (_Float16)acc;
    }
    // c4[r,s,t] = m2r_ij m2s_ik m2t_jk  (tri3)
    for (int r = 0; r < 5; r++) for (int s = r; s < 5; s++) for (int t = s; t < 5; t++) {
        float acc = 0.0f;
#pragma unroll
        for (int i = 0; i < 3; i++)
#pragma unroll
            for (int j = 0; j < 3; j++) {
                float mr = my[15 + r*9 + i*3 + j];
#pragma unroll
                for (int kq = 0; kq < 3; kq++)
                    acc += mr * my[15 + s*9 + i*3 + kq] * my[15 + t*9 + j*3 + kq];
            }
        out[f++] = (_Float16)acc;
    }
    // c5[r,s,t] = m1r_i m1s_j m2t_ij   (tri2 x 5)
    for (int r = 0; r < 5; r++) for (int s = r; s < 5; s++) for (int t = 0; t < 5; t++) {
        float acc = 0.0f;
#pragma unroll
        for (int i = 0; i < 3; i++)
#pragma unroll
            for (int j = 0; j < 3; j++)
                acc += my[r*3+i] * my[s*3+j] * my[15 + t*9 + i*3 + j];
        out[f++] = (_Float16)acc;
    }
    // c6[r,s,t] = m3r_ijk m3s_ijl m2t_kl  via T[k,l] = sum_ij m3r_ijk m3s_ijl
    for (int r = 0; r < 5; r++) for (int s = r; s < 5; s++) {
        float T[9];
#pragma unroll
        for (int e = 0; e < 9; e++) T[e] = 0.0f;
#pragma unroll
        for (int i = 0; i < 3; i++)
#pragma unroll
            for (int j = 0; j < 3; j++)
#pragma unroll
                for (int kq = 0; kq < 3; kq++) {
                    float mr = my[60 + r*27 + i*9 + j*3 + kq];
#pragma unroll
                    for (int l = 0; l < 3; l++)
                        T[kq*3 + l] += mr * my[60 + s*27 + i*9 + j*3 + l];
                }
        for (int t = 0; t < 5; t++) {
            float acc = 0.0f;
#pragma unroll
            for (int e = 0; e < 9; e++) acc += T[e] * my[15 + t*9 + e];
            out[f++] = (_Float16)acc;
        }
    }
    // c7[r,s,t] = m3r_ijk m2s_ij m1t_k  via v[k] = sum_ij m3r_ijk m2s_ij
    for (int r = 0; r < 5; r++) for (int s = 0; s < 5; s++) {
        float v0 = 0.0f, v1 = 0.0f, v2 = 0.0f;
#pragma unroll
        for (int i = 0; i < 3; i++)
#pragma unroll
            for (int j = 0; j < 3; j++) {
                float m2s = my[15 + s*9 + i*3 + j];
                v0 += my[60 + r*27 + i*9 + j*3 + 0] * m2s;
                v1 += my[60 + r*27 + i*9 + j*3 + 1] * m2s;
                v2 += my[60 + r*27 + i*9 + j*3 + 2] * m2s;
            }
        for (int t = 0; t < 5; t++) {
            float acc = v0*my[t*3+0] + v1*my[t*3+1] + v2*my[t*3+2];
            out[f++] = (_Float16)acc;
        }
    }
    // zero pad 360..383
    for (; f < KP1; f++) out[f] = (_Float16)0.0f;
}

// -------- Weight pre-pack into WMMA B-fragment lane order (f16) ---------
// fragment element i of lane L in k-step ks of n-tile nt holds
// W[ks*32 + (L>>4)*8 + (i&7) + (i>>3)*16][nt*16 + (L&15)]
__global__ void pack_w(const float* __restrict__ W, _Float16* __restrict__ out,
                       int Kact, int nk) {
    int t = blockIdx.x * blockDim.x + threadIdx.x;
    int total = nk * 16384;                          // 32 ntiles * 32 lanes * 16
    if (t >= total) return;
    int i     = t & 15;
    int lane  = (t >> 4) & 31;
    int rest  = t >> 9;                              // ntile*nk + ks
    int ks    = rest % nk;
    int ntile = rest / nk;
    int n = ntile * 16 + (lane & 15);
    int k = ks * 32 + (lane >> 4) * 8 + (i & 7) + (i >> 3) * 16;
    float v = (k < Kact) ? W[k * NU + n] : 0.0f;
    out[t] = (_Float16)v;
}

// -------- Phase 3: GEMM + bias + swish via v_wmma_f32_16x16x32_f16 ------
__global__ void gemm_swish(const _Float16* __restrict__ X, const _Float16* __restrict__ Wp,
                           const float* __restrict__ bias, _Float16* __restrict__ H,
                           int ldX, int nk) {
    int lane = threadIdx.x & 31;
    int wave = threadIdx.x >> 5;
    int tile = blockIdx.x * 8 + wave;                // 1875*32 = 60000 tiles
    int tileM = tile >> 5;
    int tileN = tile & 31;
    int half  = lane >> 4;
    int mrow  = tileM * 16 + (lane & 15);

    const _Float16* pA = X + (size_t)mrow * ldX + half * 8;
    const _Float16* pB = Wp + ((size_t)(tileN * nk) * 32 + lane) * 16;

    v8f acc = {};
    for (int ks = 0; ks < nk; ks++) {
        union { v16h v; uint4 q[2]; } a, b;
        // A frag: lane half 0 -> K {0..7,16..23}, half 1 -> K {8..15,24..31}
        a.q[0] = *reinterpret_cast<const uint4*>(pA);
        a.q[1] = *reinterpret_cast<const uint4*>(pA + 16);
        // B frag: 16 contiguous pre-packed halves per lane
        b.q[0] = *reinterpret_cast<const uint4*>(pB);
        b.q[1] = *reinterpret_cast<const uint4*>(pB + 8);
        acc = __builtin_amdgcn_wmma_f32_16x16x32_f16(
            false, a.v, false, b.v, (short)0, acc, false, false);
        pA += 32;
        pB += 32 * 16;
    }
    int col = tileN * 16 + (lane & 15);
    float bcol = bias[col];
    int rbase = tileM * 16 + half * 8;
#pragma unroll
    for (int r = 0; r < 8; r++) {
        float x = acc[r] + bcol;
        float h = x / (1.0f + __expf(-x));           // swish
        H[(size_t)(rbase + r) * NU + col] = (_Float16)h;
    }
}

// -------- Final 512->1 projection + per-species scale/shift -------------
__global__ void head_kernel(const _Float16* __restrict__ H2, const float* __restrict__ w3,
                            const float* __restrict__ b3, const int* __restrict__ Z,
                            const float* __restrict__ scale, const float* __restrict__ shift,
                            float* __restrict__ out) {
    int lane = threadIdx.x & 31;
    int wave = threadIdx.x >> 5;
    int a = blockIdx.x * 8 + wave;
    if (a >= A_ATOMS) return;
    const _Float16* h = H2 + (size_t)a * NU + lane * 16;
    const float* w = w3 + lane * 16;
    float s = 0.0f;
#pragma unroll
    for (int i = 0; i < 16; i++) s += (float)h[i] * w[i];
#pragma unroll
    for (int off = 16; off >= 1; off >>= 1) s += __shfl_xor(s, off, 32);
    if (lane == 0) {
        float v = s + b3[0];
        int z = Z[a];
        out[a] = scale[z] * v + shift[z];
    }
}

extern "C" void kernel_launch(void* const* d_in, const int* in_sizes, int n_in,
                              void* d_out, int out_size, void* d_ws, size_t ws_size,
                              hipStream_t stream) {
    const float* R     = (const float*)d_in[0];
    const int*   Z     = (const int*)  d_in[1];
    const int*   idx   = (const int*)  d_in[2];
    const float* Wr    = (const float*)d_in[3];
    const float* w1    = (const float*)d_in[4];
    const float* b1    = (const float*)d_in[5];
    const float* w2    = (const float*)d_in[6];
    const float* b2    = (const float*)d_in[7];
    const float* w3    = (const float*)d_in[8];
    const float* b3    = (const float*)d_in[9];
    const float* scale = (const float*)d_in[10];
    const float* shift = (const float*)d_in[11];

    char* ws = (char*)d_ws;
    float*    mom = (float*)   (ws + 0);           // 30000*100*4  = 12,000,000
    _Float16* gm  = (_Float16*)(ws + 12000000);    // 30000*384*2  = 23,040,000
    _Float16* w1p = (_Float16*)(ws + 35040000);    // 384*512*2    =    393,216
    _Float16* w2p = (_Float16*)(ws + 35433216);    // 512*512*2    =    524,288
    _Float16* h1  = (_Float16*)(ws + 35957504);    // 30000*512*2  = 30,720,000
    _Float16* h2  = (_Float16*)(ws + 66677504);    // 30000*512*2  = 30,720,000

    // zero per-atom moment accumulators (fresh every call)
    zero_f32<<<(A_ATOMS * MOMSZ + 255) / 256, 256, 0, stream>>>(mom, A_ATOMS * MOMSZ);
    // phase 1: pair scatter
    pair_kernel<<<(NPAIRS + 255) / 256, 256, 0, stream>>>(R, Z, idx, Wr, mom);
    // phase 2: contractions -> f16 features (padded to 384)
    contract_kernel<<<(A_ATOMS + CBLK - 1) / CBLK, CBLK, 0, stream>>>(mom, gm);
    // pack weights into WMMA fragment order
    pack_w<<<(12 * 16384 + 255) / 256, 256, 0, stream>>>(w1, w1p, NFEAT, 12);
    pack_w<<<(16 * 16384 + 255) / 256, 256, 0, stream>>>(w2, w2p, NU, 16);
    // phase 3: MLP via WMMA
    gemm_swish<<<7500, 256, 0, stream>>>(gm, w1p, b1, h1, KP1, 12);
    gemm_swish<<<7500, 256, 0, stream>>>(h1, w2p, b2, h2, NU, 16);
    head_kernel<<<3750, 256, 0, stream>>>(h2, w3, b3, Z, scale, shift, (float*)d_out);
}
